// LearntPrototypes_81449759801816
// MI455X (gfx1250) — matrix operand order repaired and checked by reading, not compiled, not run on hardware
//
#include <hip/hip_runtime.h>
#include <math.h>

typedef _Float16 v16h __attribute__((ext_vector_type(16)));
typedef _Float16 v8h  __attribute__((ext_vector_type(8)));
typedef float    v8f  __attribute__((ext_vector_type(8)));
typedef unsigned int u32x4 __attribute__((ext_vector_type(4)));
typedef int          i32x8 __attribute__((ext_vector_type(8)));
typedef int          i32x4 __attribute__((ext_vector_type(4)));

#define DIM   128
#define NPROT 256
#define EPS_F 1e-12f
#define HBYTES (16 * NPROT * 8 * (int)sizeof(_Float16))   // 65536 B of f16 protos

// ---------------------------------------------------------------------------
// Kernel 1 (once): prototypes fp32 -> f16 in the chunked LDS layout
//   hproto[(c*256 + p)*8 + j] = (f16) protos[p*128 + c*8 + j]
// plus per-prototype squared norms p2[256] (exact, fp32).
// ---------------------------------------------------------------------------
__global__ __launch_bounds__(NPROT) void proto_prep_kernel(
    const float* __restrict__ protos,
    _Float16* __restrict__ hproto,
    float* __restrict__ p2)
{
    int p = threadIdx.x;                       // 256 threads, 1 block
    const float4* row = (const float4*)(protos + (size_t)p * DIM);
    float s = 0.f;
#pragma unroll
    for (int c = 0; c < 16; ++c) {             // chunk c covers d = c*8 .. c*8+7
        float4 q0 = row[c * 2 + 0];
        float4 q1 = row[c * 2 + 1];
        s += q0.x * q0.x + q0.y * q0.y + q0.z * q0.z + q0.w * q0.w;
        s += q1.x * q1.x + q1.y * q1.y + q1.z * q1.z + q1.w * q1.w;
        v8h h;
        h[0] = (_Float16)q0.x; h[1] = (_Float16)q0.y;
        h[2] = (_Float16)q0.z; h[3] = (_Float16)q0.w;
        h[4] = (_Float16)q1.x; h[5] = (_Float16)q1.y;
        h[6] = (_Float16)q1.z; h[7] = (_Float16)q1.w;
        *(v8h*)(hproto + ((size_t)c * NPROT + p) * 8) = h;
    }
    p2[p] = s;
}

// ---------------------------------------------------------------------------
// Kernel 2: tiled cdist via f16 WMMA (f32 accumulate).
//   block = 256 threads = 8 waves; wave w handles rows [tile*16, tile*16+16)
//   B staged into LDS (64 KB) by the Tensor Data Mover (one descriptor).
// ---------------------------------------------------------------------------
__global__ __launch_bounds__(256) void cdist_wmma_kernel(
    const float* __restrict__ emb,
    const _Float16* __restrict__ hproto,   // 64 KB, chunked layout, L2-hot
    const float* __restrict__ p2g,
    float* __restrict__ out,
    int nTiles)
{
    extern __shared__ _Float16 sB[];   // 16 chunks * 256 protos * 8 halves = 64 KB

    const int wave = threadIdx.x >> 5;
    const int lane = threadIdx.x & 31;

    // --- TDM: DMA the 64 KB f16 prototype image into LDS (wave 0 only) ----
    if (wave == 0) {
        unsigned long long ga = (unsigned long long)hproto;
        // D# group 0: count=1, lds_addr=0 (dynamic LDS base), global addr, type=2
        u32x4 g0 = { 1u,                                   // count=1, user mode
                     0u,                                   // lds_addr = 0
                     (unsigned)(ga & 0xffffffffu),         // global_addr[31:0]
                     (unsigned)((ga >> 32) & 0x01ffffffu)  // global_addr[56:32]
                         | (2u << 30) };                   // type = 2 ("image")
        // D# group 1: mask=0, data_size=1 (2B), dim0=tile0=32768, dim1=tile1=1
        i32x8 g1 = { (int)0x00010000,   // data_size = 2 bytes
                     (int)0x80000000,   // tensor_dim0[15:0] = 32768 (bits 63:48)
                     (int)0x00010000,   // tensor_dim0 hi = 0, tensor_dim1 = 1
                     (int)0x80000000,   // tile_dim0 = 32768 (bits 127:112)
                     1,                 // tile_dim1 = 1
                     0x8000,            // tensor_dim0_stride = 32768
                     0, 0 };
        i32x4 gz4 = { 0, 0, 0, 0 };     // groups 2/3: dims 2+ unused
        i32x8 gz8 = { 0, 0, 0, 0, 0, 0, 0, 0 };
        __builtin_amdgcn_tensor_load_to_lds(g0, g1, gz4, gz4, gz8, 0);
        __builtin_amdgcn_s_wait_tensorcnt(0);
    }
    __syncthreads();

    const int tile = blockIdx.x * 8 + wave;
    if (tile >= nTiles) return;        // uniform per wave; EXEC stays all-1s

    const int  row   = lane & 15;
    const int  hilo  = lane >> 4;      // 0: lanes 0-15, 1: lanes 16-31
    const int  koff  = hilo * 8;       // A-fragment K sub-offset per ISA layout
    const long rbase = (long)tile * 16;

    // --- load + convert the 16x128 A tile; fold in ||e||^2 on the fly ----
    const float* arow = emb + (rbase + row) * (long)DIM;
    float e2p = 0.f;
    v16h a[4];
#pragma unroll
    for (int kb = 0; kb < 4; ++kb) {
        const int k0 = kb * 32 + koff;           // halves 0-7  : K = k0..k0+7
        const int k1 = k0 + 16;                  // halves 8-15 : K = k1..k1+7
        float4 q0 = *(const float4*)(arow + k0);
        float4 q1 = *(const float4*)(arow + k0 + 4);
        float4 q2 = *(const float4*)(arow + k1);
        float4 q3 = *(const float4*)(arow + k1 + 4);
        e2p += q0.x * q0.x + q0.y * q0.y + q0.z * q0.z + q0.w * q0.w;
        e2p += q1.x * q1.x + q1.y * q1.y + q1.z * q1.z + q1.w * q1.w;
        e2p += q2.x * q2.x + q2.y * q2.y + q2.z * q2.z + q2.w * q2.w;
        e2p += q3.x * q3.x + q3.y * q3.y + q3.z * q3.z + q3.w * q3.w;
        v16h t;
        t[0]  = (_Float16)q0.x; t[1]  = (_Float16)q0.y;
        t[2]  = (_Float16)q0.z; t[3]  = (_Float16)q0.w;
        t[4]  = (_Float16)q1.x; t[5]  = (_Float16)q1.y;
        t[6]  = (_Float16)q1.z; t[7]  = (_Float16)q1.w;
        t[8]  = (_Float16)q2.x; t[9]  = (_Float16)q2.y;
        t[10] = (_Float16)q2.z; t[11] = (_Float16)q2.w;
        t[12] = (_Float16)q3.x; t[13] = (_Float16)q3.y;
        t[14] = (_Float16)q3.z; t[15] = (_Float16)q3.w;
        a[kb] = t;
    }
    // lanes L and L^16 each hold half of row (L&15)'s norm -> combine
    const float e2full = e2p + __shfl_xor(e2p, 16, 32); // lane L: ||e[row]||^2

    // hoist the convergent cross-lane reads out of the column-tile loop:
    // C-layout VGPR i -> row i (lanes 0-15) / row 8+i (lanes 16-31)
    float e2s[8];
#pragma unroll
    for (int i = 0; i < 8; ++i)
        e2s[i] = __shfl(e2full, (lane & 16) ? (8 + i) : i, 32);

    // --- 16 column tiles of prototypes ------------------------------------
#pragma unroll
    for (int ct = 0; ct < 16; ++ct) {
        const int pbase = ct * 16;
        v8f acc = {};
#pragma unroll
        for (int kb = 0; kb < 4; ++kb) {
            // B layout (dense f16 32x16): col = lane&15,
            // element e<8 holds K = kb*32 + hilo*8 + e, e>=8 holds K = kb*32 + 16 + hilo*8 + (e-8) -> chunks ch0, ch0+2
            const int ch0 = kb * 4 + hilo;
            const _Float16* bp =
                sB + ((size_t)ch0 * NPROT + pbase + (lane & 15)) * 8;
            v8h lo = *(const v8h*)(bp);
            v8h hi = *(const v8h*)(bp + (size_t)2 * NPROT * 8);
            v16h b = __builtin_shufflevector(
                lo, hi, 0, 1, 2, 3, 4, 5, 6, 7, 8, 9, 10, 11, 12, 13, 14, 15);
            acc = __builtin_amdgcn_wmma_f32_16x16x32_f16(
                false, a[kb], false, b, (short)0, acc, false, false);
        }

        const float p2v = p2g[pbase + (lane & 15)];   // L2-hot
#pragma unroll
        for (int i = 0; i < 8; ++i) {
            const int rsel = hilo ? (8 + i) : i;
            float d2 = e2s[i] + p2v - 2.0f * acc[i];
            d2 = fmaxf(d2, EPS_F);
            out[(rbase + rsel) * (long)NPROT + pbase + (lane & 15)] =
                -sqrtf(d2);
        }
    }
}

// ---------------------------------------------------------------------------
extern "C" void kernel_launch(void* const* d_in, const int* in_sizes, int n_in,
                              void* d_out, int out_size, void* d_ws, size_t ws_size,
                              hipStream_t stream)
{
    const float* emb    = (const float*)d_in[0];   // (N, 128) f32
    const float* protos = (const float*)d_in[1];   // (256, 128) f32
    float*       out    = (float*)d_out;           // (N, 1, 256) f32

    // workspace layout: [0, 64KB) f16 prototypes (chunked), then 256 f32 norms
    _Float16* hproto = (_Float16*)d_ws;
    float*    p2     = (float*)((char*)d_ws + HBYTES);

    const int N      = in_sizes[0] / DIM;          // 500000
    const int nTiles = (N + 15) / 16;              // 31250 (exact)
    const int nBlks  = (nTiles + 7) / 8;

    proto_prep_kernel<<<1, NPROT, 0, stream>>>(protos, hproto, p2);
    cdist_wmma_kernel<<<nBlks, 256, HBYTES, stream>>>(emb, hproto, p2, out, nTiles);
}
